// MambaLOBModel_34437047779590
// MI455X (gfx1250) — matrix operand order repaired and checked by reading, not compiled, not run on hardware
//
#include <hip/hip_runtime.h>
#include <hip/hip_bf16.h>

// ---------------- model dims ----------------
#define BQ      8
#define LQ      2048
#define DINQ    40
#define HQ      128
#define NLQ     4
#define DSTATE  16
#define DCONV   4
#define DTRANK  8
#define DINNER  256
#define NHOR    3
#define NCLS    3
#define ROWS    (BQ * LQ)              // 16384 tokens
#define XPN     (DTRANK + 2 * DSTATE)  // 40
#define CL      32                     // scan chunk length
#define NC      (LQ / CL)              // 64 chunks

typedef __attribute__((ext_vector_type(16))) __bf16 v16bf;
typedef __attribute__((ext_vector_type(8)))  float  v8f;

// ---------------- small math helpers ----------------
__device__ __forceinline__ float silu_f(float x) { return x / (1.f + __expf(-x)); }
__device__ __forceinline__ float softplus_f(float x) {
  return (x > 20.f) ? x : log1pf(expf(x));
}
__device__ __forceinline__ float gelu_f(float x) {
  return 0.5f * x * (1.f + erff(x * 0.70710678118654752f));
}

// ---------------- WMMA fragment loaders (fp32 mem -> bf16 regs) ----------------
// A: row-major MxK fp32, 16x32 tile, contiguous k order (k = 16*half + s, matches B)
__device__ __forceinline__ v16bf load_a_frag(const float* A, int lda, int m0, int k0, int lane) {
  const int half = lane >> 4;
  const float* row = A + (size_t)(m0 + (lane & 15)) * lda;
  v16bf a;
#pragma unroll
  for (int i = 0; i < 8; ++i) {
    int kb = k0 + half * 16 + i * 2;  // half*16+i*2 (contiguous, matches B)
    a[2 * i]     = (__bf16)row[kb];
    a[2 * i + 1] = (__bf16)row[kb + 1];
  }
  return a;
}
// B = W^T where W is row-major (N,K) fp32.  B[k][n] = W[n*ldw + k].
// 16-bit dense B 32x16: lanes 0-15 hold K=k0..k0+15 of column n, lanes 16-31 K=k0+16..k0+31.
__device__ __forceinline__ v16bf load_b_fragT(const float* W, int ldw, int n0, int k0,
                                              int nmax, int lane) {
  int n = n0 + (lane & 15);
  if (n >= nmax) n = nmax - 1;  // clamp; cols >= nmax never stored
  const float* p = W + (size_t)n * ldw + k0 + ((lane >> 4) << 4);
  v16bf b;
#pragma unroll
  for (int j = 0; j < 16; ++j) b[j] = (__bf16)p[j];
  return b;
}

// ---------------- GEMM: D[M,N] = A[M,K] @ W[N,K]^T ----------------
// One wave computes a 16 x (16*NW) strip: A fragment loaded once per K-step,
// NW back-to-back WMMAs into independent accumulators.
template <int NW>
__global__ __launch_bounds__(256)
void k_gemm(const float* __restrict__ A, const float* __restrict__ W,
            float* __restrict__ D, int M, int N, int K, int ngroups, int nmax) {
  int wid  = (blockIdx.x * blockDim.x + threadIdx.x) >> 5;
  int lane = threadIdx.x & 31;
  int tiles_m = M >> 4;
  if (wid >= tiles_m * ngroups) return;  // wave-uniform
  int tm = wid / ngroups, tg = wid % ngroups;
  v8f c[NW];
#pragma unroll
  for (int w = 0; w < NW; ++w) c[w] = (v8f){};
  for (int k0 = 0; k0 < K; k0 += 32) {
    v16bf a = load_a_frag(A, K, tm * 16, k0, lane);
#pragma unroll
    for (int w = 0; w < NW; ++w) {
      v16bf b = load_b_fragT(W, K, (tg * NW + w) * 16, k0, nmax, lane);
      c[w] = __builtin_amdgcn_wmma_f32_16x16x32_bf16(false, a, false, b, (short)0, c[w],
                                                     false, false);
    }
  }
  int mb = tm * 16 + ((lane >> 4) << 3);
#pragma unroll
  for (int w = 0; w < NW; ++w) {
    int n = (tg * NW + w) * 16 + (lane & 15);
    if (n < nmax) {
#pragma unroll
      for (int r = 0; r < 8; ++r) D[(size_t)(mb + r) * N + n] = c[w][r];
    }
  }
}

// out_proj with fused residual accumulate into H (C preloaded from H, stored back)
template <int NW>
__global__ __launch_bounds__(256)
void k_gemm_res(const float* __restrict__ A, const float* __restrict__ W,
                float* __restrict__ Hio, int M, int N, int K, int ngroups) {
  int wid  = (blockIdx.x * blockDim.x + threadIdx.x) >> 5;
  int lane = threadIdx.x & 31;
  int tiles_m = M >> 4;
  if (wid >= tiles_m * ngroups) return;
  int tm = wid / ngroups, tg = wid % ngroups;
  int mb = tm * 16 + ((lane >> 4) << 3);
  v8f c[NW];
#pragma unroll
  for (int w = 0; w < NW; ++w) {
    int n = (tg * NW + w) * 16 + (lane & 15);
#pragma unroll
    for (int r = 0; r < 8; ++r) c[w][r] = Hio[(size_t)(mb + r) * N + n];
  }
  for (int k0 = 0; k0 < K; k0 += 32) {
    v16bf a = load_a_frag(A, K, tm * 16, k0, lane);
#pragma unroll
    for (int w = 0; w < NW; ++w) {
      v16bf b = load_b_fragT(W, K, (tg * NW + w) * 16, k0, N, lane);
      c[w] = __builtin_amdgcn_wmma_f32_16x16x32_bf16(false, a, false, b, (short)0, c[w],
                                                     false, false);
    }
  }
#pragma unroll
  for (int w = 0; w < NW; ++w) {
    int n = (tg * NW + w) * 16 + (lane & 15);
#pragma unroll
    for (int r = 0; r < 8; ++r) Hio[(size_t)(mb + r) * N + n] = c[w][r];
  }
}

// ---------------- input projection (K=40) + LayerNorm, fused ----------------
__global__ __launch_bounds__(128)
void k_input_proj_ln(const float* __restrict__ x, const float* __restrict__ w,
                     const float* __restrict__ bias, const float* __restrict__ g,
                     const float* __restrict__ bt, float* __restrict__ out) {
  int row = blockIdx.x;  // b*L + l
  int j   = threadIdx.x; // 0..127
  __shared__ float xr[DINQ];
  __shared__ float s1[HQ], s2[HQ];
  if (j < DINQ) xr[j] = x[(size_t)row * DINQ + j];
  __syncthreads();
  float s = bias[j];
  const float* wr = w + (size_t)j * DINQ;
#pragma unroll
  for (int k = 0; k < DINQ; ++k) s += xr[k] * wr[k];
  s1[j] = s; s2[j] = s * s;
  __syncthreads();
  for (int off = HQ / 2; off > 0; off >>= 1) {
    if (j < off) { s1[j] += s1[j + off]; s2[j] += s2[j + off]; }
    __syncthreads();
  }
  float m = s1[0] * (1.f / HQ);
  float v = s2[0] * (1.f / HQ) - m * m;
  out[(size_t)row * HQ + j] = (s - m) * rsqrtf(v + 1e-5f) * g[j] + bt[j];
}

// ---------------- LayerNorm over H=128 ----------------
__global__ __launch_bounds__(128)
void k_ln(const float* __restrict__ in, const float* __restrict__ g,
          const float* __restrict__ bt, float* __restrict__ out) {
  int row = blockIdx.x, j = threadIdx.x;
  __shared__ float s1[HQ], s2[HQ];
  float v = in[(size_t)row * HQ + j];
  s1[j] = v; s2[j] = v * v;
  __syncthreads();
  for (int off = HQ / 2; off > 0; off >>= 1) {
    if (j < off) { s1[j] += s1[j + off]; s2[j] += s2[j + off]; }
    __syncthreads();
  }
  float m  = s1[0] * (1.f / HQ);
  float va = s2[0] * (1.f / HQ) - m * m;
  out[(size_t)row * HQ + j] = (v - m) * rsqrtf(va + 1e-5f) * g[j] + bt[j];
}

// ---------------- depthwise causal conv4 + SiLU (reads xh half of xz) ----------------
__global__ __launch_bounds__(256)
void k_conv_silu(const float* __restrict__ xz, const float* __restrict__ cw,
                 const float* __restrict__ cb, float* __restrict__ xhact) {
  int idx = blockIdx.x * blockDim.x + threadIdx.x;
  if (idx >= ROWS * DINNER) return;
  int d = idx % DINNER, row = idx / DINNER, l = row % LQ;
  float s = cb[d];
#pragma unroll
  for (int jj = 0; jj < DCONV; ++jj) {
    int ls = l - (DCONV - 1) + jj;
    if (ls >= 0)
      s += cw[d * DCONV + jj] * xz[(size_t)(row - (DCONV - 1) + jj) * (2 * DINNER) + d];
  }
  xhact[idx] = silu_f(s);
}

// ---------------- delta = softplus(dt_in @ dt_w^T + dt_b) ----------------
__global__ __launch_bounds__(256)
void k_delta(const float* __restrict__ xp, const float* __restrict__ dtw,
             const float* __restrict__ dtb, float* __restrict__ delta) {
  int idx = blockIdx.x * blockDim.x + threadIdx.x;
  if (idx >= ROWS * DINNER) return;
  int d = idx % DINNER, row = idx / DINNER;
  const float* xr = xp + (size_t)row * XPN;
  float s = dtb[d];
#pragma unroll
  for (int r = 0; r < DTRANK; ++r) s += xr[r] * dtw[d * DTRANK + r];
  delta[idx] = softplus_f(s);
}

// ---------------- chunked selective scan ----------------
// Summary buffers laid out [b][chunk][n][d] so d is lane-contiguous everywhere.
#define SUMIDX(b, ck, n, d) ((((size_t)(b) * NC + (ck)) * DSTATE + (n)) * DINNER + (d))

// phase 1: per (b,ck,d): cumulative a-product + partial state (h_in = 0)
__global__ __launch_bounds__(256)
void k_scan1(const float* __restrict__ delta, const float* __restrict__ xp,
             const float* __restrict__ xhact, const float* __restrict__ Alog,
             float* __restrict__ Aprod, float* __restrict__ Hpart) {
  int t = blockIdx.x * blockDim.x + threadIdx.x;
  if (t >= BQ * DINNER * NC) return;
  int d  = t % DINNER;
  int ck = (t / DINNER) % NC;
  int b  = t / (DINNER * NC);
  float an[DSTATE], Ap[DSTATE], Hp[DSTATE];
#pragma unroll
  for (int n = 0; n < DSTATE; ++n) { an[n] = -__expf(Alog[n]); Ap[n] = 1.f; Hp[n] = 0.f; }
  int l0 = ck * CL;
  for (int tt = 0; tt < CL; ++tt) {
    int row = b * LQ + l0 + tt;
    float dt = delta[(size_t)row * DINNER + d];
    float xv = xhact[(size_t)row * DINNER + d];
    const float* xr = xp + (size_t)row * XPN;
#pragma unroll
    for (int n = 0; n < DSTATE; ++n) {
      float da = __expf(dt * an[n]);
      Hp[n] = da * Hp[n] + dt * xr[DTRANK + n] * xv;
      Ap[n] *= da;
    }
  }
#pragma unroll
  for (int n = 0; n < DSTATE; ++n) {
    Aprod[SUMIDX(b, ck, n, d)] = Ap[n];
    Hpart[SUMIDX(b, ck, n, d)] = Hp[n];
  }
}

// phase 2: prefix over chunk summaries -> carry-in state per chunk
__global__ __launch_bounds__(256)
void k_scan2(const float* __restrict__ Aprod, const float* __restrict__ Hpart,
             float* __restrict__ carry) {
  int t = blockIdx.x * blockDim.x + threadIdx.x;
  if (t >= BQ * DINNER) return;
  int d = t % DINNER, b = t / DINNER;
  float c[DSTATE];
#pragma unroll
  for (int n = 0; n < DSTATE; ++n) c[n] = 0.f;
  for (int ck = 0; ck < NC; ++ck) {
#pragma unroll
    for (int n = 0; n < DSTATE; ++n) {
      size_t ix = SUMIDX(b, ck, n, d);
      carry[ix] = c[n];
      c[n] = Aprod[ix] * c[n] + Hpart[ix];
    }
  }
}

// phase 3: replay chunk with carry; fuse y = h.C + D*xh, * silu(z)
__global__ __launch_bounds__(256)
void k_scan3(const float* __restrict__ delta, const float* __restrict__ xp,
             const float* __restrict__ xhact, const float* __restrict__ xz,
             const float* __restrict__ Alog, const float* __restrict__ Dp,
             const float* __restrict__ carry, float* __restrict__ ybuf) {
  int t = blockIdx.x * blockDim.x + threadIdx.x;
  if (t >= BQ * DINNER * NC) return;
  int d  = t % DINNER;
  int ck = (t / DINNER) % NC;
  int b  = t / (DINNER * NC);
  float an[DSTATE], hc[DSTATE];
#pragma unroll
  for (int n = 0; n < DSTATE; ++n) {
    an[n] = -__expf(Alog[n]);
    hc[n] = carry[SUMIDX(b, ck, n, d)];
  }
  float dcoef = Dp[d];
  int l0 = ck * CL;
  for (int tt = 0; tt < CL; ++tt) {
    int row = b * LQ + l0 + tt;
    float dt = delta[(size_t)row * DINNER + d];
    float xv = xhact[(size_t)row * DINNER + d];
    const float* xr = xp + (size_t)row * XPN;
    float y = 0.f;
#pragma unroll
    for (int n = 0; n < DSTATE; ++n) {
      float da = __expf(dt * an[n]);
      hc[n] = da * hc[n] + dt * xr[DTRANK + n] * xv;
      y += hc[n] * xr[DTRANK + DSTATE + n];
    }
    float zv = xz[(size_t)row * (2 * DINNER) + DINNER + d];
    ybuf[(size_t)row * DINNER + d] = (y + dcoef * xv) * silu_f(zv);
  }
}

// ---------------- classification heads on feats = h[:, L-1, :] ----------------
__global__ __launch_bounds__(256)
void k_heads(const float* __restrict__ h, const float* __restrict__ w1,
             const float* __restrict__ b1, const float* __restrict__ w2,
             const float* __restrict__ b2, float* __restrict__ out) {
  __shared__ float t1[BQ][HQ / 2];
  for (int hor = 0; hor < NHOR; ++hor) {
    for (int idx = threadIdx.x; idx < BQ * (HQ / 2); idx += blockDim.x) {
      int b = idx / (HQ / 2), j = idx % (HQ / 2);
      const float* feat = h + ((size_t)(b * LQ + (LQ - 1))) * HQ;
      const float* wr = w1 + ((size_t)hor * (HQ / 2) + j) * HQ;
      float s = b1[hor * (HQ / 2) + j];
#pragma unroll
      for (int k = 0; k < HQ; ++k) s += feat[k] * wr[k];
      t1[b][j] = gelu_f(s);
    }
    __syncthreads();
    for (int idx = threadIdx.x; idx < BQ * NCLS; idx += blockDim.x) {
      int b = idx / NCLS, c = idx % NCLS;
      const float* wr = w2 + ((size_t)hor * NCLS + c) * (HQ / 2);
      float s = b2[hor * NCLS + c];
#pragma unroll
      for (int j = 0; j < HQ / 2; ++j) s += t1[b][j] * wr[j];
      out[((size_t)b * NHOR + hor) * NCLS + c] = s;
    }
    __syncthreads();
  }
}

// ---------------- host launch ----------------
extern "C" void kernel_launch(void* const* d_in, const int* in_sizes, int n_in,
                              void* d_out, int out_size, void* d_ws, size_t ws_size,
                              hipStream_t stream) {
  const float* x    = (const float*)d_in[0];
  const float* ipw  = (const float*)d_in[1];
  const float* ipb  = (const float*)d_in[2];
  const float* ing  = (const float*)d_in[3];
  const float* inb  = (const float*)d_in[4];
  const float* ng   = (const float*)d_in[5];
  const float* nb   = (const float*)d_in[6];
  const float* ipjw = (const float*)d_in[7];   // (NL, 512, 128)
  const float* cw   = (const float*)d_in[8];   // (NL, 256, 1, 4)
  const float* cb   = (const float*)d_in[9];
  const float* xpw  = (const float*)d_in[10];  // (NL, 40, 256)
  const float* dtw  = (const float*)d_in[11];  // (NL, 256, 8)
  const float* dtb  = (const float*)d_in[12];
  const float* Alog = (const float*)d_in[13];  // (NL, 16)
  const float* Dp   = (const float*)d_in[14];  // (NL, 256)
  const float* opw  = (const float*)d_in[15];  // (NL, 128, 256)
  const float* hw1  = (const float*)d_in[16];
  const float* hb1  = (const float*)d_in[17];
  const float* hw2  = (const float*)d_in[18];
  const float* hb2  = (const float*)d_in[19];
  float* out = (float*)d_out;

  // workspace layout (floats)
  float* ws    = (float*)d_ws;
  float* h     = ws;                          // ROWS*H
  float* hn    = h     + (size_t)ROWS * HQ;
  float* xz    = hn    + (size_t)ROWS * HQ;   // ROWS*512
  float* xhact = xz    + (size_t)ROWS * 2 * DINNER;
  float* xp    = xhact + (size_t)ROWS * DINNER;  // ROWS*40
  float* delta = xp    + (size_t)ROWS * XPN;
  float* ybuf  = delta + (size_t)ROWS * DINNER;
  float* Aprod = ybuf  + (size_t)ROWS * DINNER;  // B*NC*DSTATE*DINNER
  float* Hpart = Aprod + (size_t)BQ * NC * DSTATE * DINNER;
  float* carry = Hpart + (size_t)BQ * NC * DSTATE * DINNER;

  // stage 0: input projection + LN
  k_input_proj_ln<<<ROWS, 128, 0, stream>>>(x, ipw, ipb, ing, inb, h);

  for (int i = 0; i < NLQ; ++i) {
    k_ln<<<ROWS, 128, 0, stream>>>(h, ng + i * HQ, nb + i * HQ, hn);

    { // in_proj: M=16384, N=512, K=128; 16x64 strip per wave
      int ngroups = (2 * DINNER) / 64;
      int waves = (ROWS / 16) * ngroups;
      k_gemm<4><<<(waves * 32 + 255) / 256, 256, 0, stream>>>(
          hn, ipjw + (size_t)i * 2 * DINNER * HQ, xz, ROWS, 2 * DINNER, HQ, ngroups,
          2 * DINNER);
    }

    k_conv_silu<<<(ROWS * DINNER + 255) / 256, 256, 0, stream>>>(
        xz, cw + (size_t)i * DINNER * DCONV, cb + (size_t)i * DINNER, xhact);

    { // x_proj: M=16384, N=40 (one 3-subtile group, guarded), K=256
      int ngroups = 1;
      int waves = (ROWS / 16) * ngroups;
      k_gemm<3><<<(waves * 32 + 255) / 256, 256, 0, stream>>>(
          xhact, xpw + (size_t)i * XPN * DINNER, xp, ROWS, XPN, DINNER, ngroups, XPN);
    }

    k_delta<<<(ROWS * DINNER + 255) / 256, 256, 0, stream>>>(
        xp, dtw + (size_t)i * DINNER * DTRANK, dtb + (size_t)i * DINNER, delta);

    k_scan1<<<(BQ * DINNER * NC + 255) / 256, 256, 0, stream>>>(
        delta, xp, xhact, Alog + (size_t)i * DSTATE, Aprod, Hpart);
    k_scan2<<<(BQ * DINNER + 255) / 256, 256, 0, stream>>>(Aprod, Hpart, carry);
    k_scan3<<<(BQ * DINNER * NC + 255) / 256, 256, 0, stream>>>(
        delta, xp, xhact, xz, Alog + (size_t)i * DSTATE, Dp + (size_t)i * DINNER, carry,
        ybuf);

    { // out_proj + residual: M=16384, N=128, K=256; 16x64 strip per wave
      int ngroups = HQ / 64;
      int waves = (ROWS / 16) * ngroups;
      k_gemm_res<4><<<(waves * 32 + 255) / 256, 256, 0, stream>>>(
          ybuf, opw + (size_t)i * HQ * DINNER, h, ROWS, HQ, DINNER, ngroups);
    }
  }

  k_heads<<<1, 256, 0, stream>>>(h, hw1, hb1, hw2, hb2, out);
}
